// OrientationLearner_54924041781907
// MI455X (gfx1250) — matrix operand order repaired and checked by reading, not compiled, not run on hardware
//
#include <hip/hip_runtime.h>
#include <hip/hip_bf16.h>

typedef __attribute__((ext_vector_type(16))) _Float16 v16h;
typedef __attribute__((ext_vector_type(8)))  _Float16 v8h;
typedef __attribute__((ext_vector_type(8)))  float    v8f;
typedef __attribute__((ext_vector_type(4)))  float    v4f;

#define HIDDEN   128
#define EDGE_NF  16
#define K1_REAL  (EDGE_NF + 2 * HIDDEN)   // 272
#define NCHUNK1  9                        // ceil(272/32)
#define NCHUNK2  4                        // 128/32
#define FRAG_H   16                       // halfs per lane per fragment
#define W1S_ELEMS (NCHUNK1 * 8 * 32 * FRAG_H)   // 36864
#define W2S_ELEMS (NCHUNK2 * 8 * 32 * FRAG_H)   // 16384
#define WAVES_PB 8                        // 256 threads / block

__device__ __forceinline__ float silu_f(float v) {
    return v / (1.0f + __expf(-v));
}

// ---------------------------------------------------------------------------
// Prep: swizzle weights f32 -> f16 WMMA-fragment order:
//   dst[((kc*8 + t)*32 + lane)*16 + i] = W[k*128 + n],
//   k = kc*32 + (lane>>4)*8 + (i&7) + (i>>3)*16,  n = t*16 + (lane&15)   (zero-pad k >= Kreal)
// so each B fragment is one contiguous 32B load per lane (coalesced 1KB/wave).
// Also zeroes vec1/vec2 accumulators (required every call).
// ---------------------------------------------------------------------------
__global__ void prep_kernel(const float* __restrict__ eW1,
                            const float* __restrict__ eW2,
                            const float* __restrict__ v1W1,
                            const float* __restrict__ v2W1,
                            _Float16* __restrict__ eW1s,
                            _Float16* __restrict__ eW2s,
                            _Float16* __restrict__ v1W1s,
                            _Float16* __restrict__ v2W1s,
                            float* __restrict__ vec1,
                            float* __restrict__ vec2,
                            int nvec) {
    const int i = blockIdx.x * blockDim.x + threadIdx.x;
    if (i < W1S_ELEMS) {
        const int ii = i & 15;
        const int L  = (i >> 4) & 31;
        const int f  = i >> 9;
        const int t  = f & 7;
        const int kc = f >> 3;
        const int k  = kc * 32 + (L >> 4) * 8 + (ii & 7) + (ii >> 3) * 16;
        const int n  = t * 16 + (L & 15);
        eW1s[i] = (k < K1_REAL) ? (_Float16)eW1[k * HIDDEN + n] : (_Float16)0.0f;
        if (kc < NCHUNK2) {   // first 16384 entries share the same index math
            eW2s[i]  = (_Float16)eW2[k * HIDDEN + n];
            v1W1s[i] = (_Float16)v1W1[k * HIDDEN + n];
            v2W1s[i] = (_Float16)v2W1[k * HIDDEN + n];
        }
    }
    if (i < nvec) {
        vec1[i] = 0.0f;
        vec2[i] = 0.0f;
    }
}

// ---------------------------------------------------------------------------
// Edge kernel: one wave32 per 16-edge tile, 8 waves per block.
//   L1: [16x288] @ [288x128]  -> 72 v_wmma_f32_16x16x32_f16
//   L2 + 2 heads: 3 x 32 wmma; head 128->1 dots taken straight from the
//   WMMA accumulators (no LDS round-trip), reduced via a 16x16 LDS tile.
//   scatter rel_pos * w{1,2} via global f32 atomics.
// ---------------------------------------------------------------------------
__global__ void __launch_bounds__(WAVES_PB * 32)
edge_kernel(const float* __restrict__ h,
            const float* __restrict__ x,
            const int*   __restrict__ edge_index,   // [2, E]
            const float* __restrict__ edge_attr,    // [E, 16]
            const _Float16* __restrict__ eW1s, const float* __restrict__ eb1,
            const _Float16* __restrict__ eW2s, const float* __restrict__ eb2,
            const _Float16* __restrict__ v1W1s, const float* __restrict__ v1b1,
            const float* __restrict__ v1W2,     const float* __restrict__ v1b2,
            const _Float16* __restrict__ v2W1s, const float* __restrict__ v2b1,
            const float* __restrict__ v2W2,     const float* __restrict__ v2b2,
            float* __restrict__ vec1, float* __restrict__ vec2,
            int n_edges)
{
    __shared__ _Float16 lds_ef[WAVES_PB][16 * HIDDEN];  // 32 KB
    __shared__ float    lds_w [WAVES_PB][16 * 16];      //  8 KB

    const int lane = threadIdx.x & 31;
    const int wave = threadIdx.x >> 5;
    const int tile = blockIdx.x * WAVES_PB + wave;
    const int ebase = tile * 16;
    if (ebase >= n_edges) return;

    const int m    = lane & 15;
    const int half = lane >> 4;

    const int  e     = ebase + m;
    const bool valid = (e < n_edges);
    const int  ee    = valid ? e : 0;

    const int row = edge_index[ee];
    const int col = edge_index[n_edges + ee];
    const float* __restrict__ hr = h + (size_t)row * HIDDEN;
    const float* __restrict__ hc = h + (size_t)col * HIDDEN;
    const float* __restrict__ ea = edge_attr + (size_t)ee * EDGE_NF;

    // Prefetch the end-of-kernel positional gather now (global_prefetch_b8):
    // it's consumed ~170 WMMAs from here, so the HBM latency is fully hidden.
    __builtin_prefetch(x + (size_t)row * 3, 0, 3);
    __builtin_prefetch(x + (size_t)col * 3, 0, 3);

    const v16h* __restrict__ W1f = (const v16h*)eW1s;
    const v16h* __restrict__ W2f = (const v16h*)eW2s;
    const v16h* __restrict__ V1f = (const v16h*)v1W1s;
    const v16h* __restrict__ V2f = (const v16h*)v2W1s;

    const v8f zero8 = {};

    // ---------------- Layer 1 ----------------
    v8f acc[8];
#pragma unroll
    for (int t = 0; t < 8; ++t) acc[t] = zero8;

    for (int kc = 0; kc < NCHUNK1; ++kc) {
        const int kb = kc * 32;
        v16h a;
        if (kc < 8) {
            const float* src = (kc < 4) ? (hr + kb) : (hc + (kb - HIDDEN));
            const v4f f0 = *(const v4f*)(src + half * 8);
            const v4f f1 = *(const v4f*)(src + half * 8 + 4);
            const v4f f2 = *(const v4f*)(src + 16 + half * 8);
            const v4f f3 = *(const v4f*)(src + 16 + half * 8 + 4);
#pragma unroll
            for (int j = 0; j < 4; ++j) {
                a[j]      = (_Float16)f0[j];
                a[4 + j]  = (_Float16)f1[j];
                a[8 + j]  = (_Float16)f2[j];
                a[12 + j] = (_Float16)f3[j];
            }
        } else {  // edge_attr chunk (16 real K) + zero pad
            const v4f f0 = *(const v4f*)(ea + half * 8);
            const v4f f1 = *(const v4f*)(ea + half * 8 + 4);
#pragma unroll
            for (int j = 0; j < 4; ++j) {
                a[j]      = (_Float16)f0[j];
                a[4 + j]  = (_Float16)f1[j];
                a[8 + j]  = (_Float16)0.0f;
                a[12 + j] = (_Float16)0.0f;
            }
        }
#pragma unroll
        for (int t = 0; t < 8; ++t) {
            const v16h b = W1f[(size_t)(kc * 8 + t) * 32 + lane];
            acc[t] = __builtin_amdgcn_wmma_f32_16x16x32_f16(
                false, a, false, b, (short)0, acc[t], false, false);
        }
    }

    _Float16* __restrict__ EF = lds_ef[wave];

#pragma unroll
    for (int t = 0; t < 8; ++t) {
        const int n = t * 16 + m;
        const float bias = eb1[n];
#pragma unroll
        for (int r = 0; r < 8; ++r)
            EF[(r + half * 8) * HIDDEN + n] = (_Float16)silu_f(acc[t][r] + bias);
    }
    __builtin_amdgcn_wave_barrier();

    // generic [16x128] @ [128x128]: A from LDS (2x16B ds loads), B from frags
    auto gemm128 = [&](const _Float16* __restrict__ A_lds,
                       const v16h* __restrict__ Wf,
                       v8f* __restrict__ c) {
#pragma unroll
        for (int kc = 0; kc < NCHUNK2; ++kc) {
            const int kb = kc * 32;
            const v8h lo = *(const v8h*)(A_lds + m * HIDDEN + kb + half * 8);
            const v8h hi = *(const v8h*)(A_lds + m * HIDDEN + kb + 16 + half * 8);
            const v16h a = __builtin_shufflevector(lo, hi,
                0, 1, 2, 3, 4, 5, 6, 7, 8, 9, 10, 11, 12, 13, 14, 15);
#pragma unroll
            for (int t = 0; t < 8; ++t) {
                const v16h b = Wf[(size_t)(kc * 8 + t) * 32 + lane];
                c[t] = __builtin_amdgcn_wmma_f32_16x16x32_f16(
                    false, a, false, b, (short)0, c[t], false, false);
            }
        }
    };

    // head finisher, straight from accumulators:
    //   w[m] = sum_n silu(z[m][n] + b1[n]) * W2[n] + b2
    // Lane (half,lm) holds rows r -> m=r+half*8, columns n=t*16+lm.
    // 64 FMAs/lane -> 8 row partials -> 16x16 LDS tile -> lane m row-sums.
    auto head_dot = [&](v8f* accz, const float* __restrict__ b1,
                        const float* __restrict__ W2v, float b2) -> float {
        const int lm = lane & 15;
        float pr[8];
#pragma unroll
        for (int r = 0; r < 8; ++r) pr[r] = 0.0f;
#pragma unroll
        for (int t = 0; t < 8; ++t) {
            const int n = t * 16 + lm;
            const float bias = b1[n];
            const float wv = W2v[n];
#pragma unroll
            for (int r = 0; r < 8; ++r)
                pr[r] += silu_f(accz[t][r] + bias) * wv;
        }
        float* WS = lds_w[wave];
#pragma unroll
        for (int r = 0; r < 8; ++r)
            WS[(r + half * 8) * 16 + lm] = pr[r];
        __builtin_amdgcn_wave_barrier();
        float s = b2;
#pragma unroll
        for (int q = 0; q < 4; ++q) {
            const v4f v = *(const v4f*)(WS + lm * 16 + q * 4);
            s += v[0] + v[1] + v[2] + v[3];
        }
        return s;   // every lane holds w for edge m = lane&15
    };

    // ---------------- Layer 2 ----------------
    v8f acc2[8];
#pragma unroll
    for (int t = 0; t < 8; ++t) acc2[t] = zero8;
    gemm128(EF, W2f, acc2);
    __builtin_amdgcn_wave_barrier();
#pragma unroll
    for (int t = 0; t < 8; ++t) {
        const int n = t * 16 + m;
        const float bias = eb2[n];
#pragma unroll
        for (int r = 0; r < 8; ++r)
            EF[(r + half * 8) * HIDDEN + n] = (_Float16)silu_f(acc2[t][r] + bias);
    }
    __builtin_amdgcn_wave_barrier();

    // ---------------- Heads ----------------
    float w1, w2;
    {
        v8f accz[8];
#pragma unroll
        for (int t = 0; t < 8; ++t) accz[t] = zero8;
        gemm128(EF, V1f, accz);
        w1 = head_dot(accz, v1b1, v1W2, v1b2[0]);
    }
    __builtin_amdgcn_wave_barrier();
    {
        v8f accz[8];
#pragma unroll
        for (int t = 0; t < 8; ++t) accz[t] = zero8;
        gemm128(EF, V2f, accz);
        w2 = head_dot(accz, v2b1, v2W2, v2b2[0]);
    }

    // ---------------- Scatter ----------------
    if (half == 0 && valid) {
        const float rx = x[col * 3 + 0] - x[row * 3 + 0];
        const float ry = x[col * 3 + 1] - x[row * 3 + 1];
        const float rz = x[col * 3 + 2] - x[row * 3 + 2];
        atomicAdd(&vec1[row * 3 + 0], rx * w1);
        atomicAdd(&vec1[row * 3 + 1], ry * w1);
        atomicAdd(&vec1[row * 3 + 2], rz * w1);
        atomicAdd(&vec2[row * 3 + 0], rx * w2);
        atomicAdd(&vec2[row * 3 + 1], ry * w2);
        atomicAdd(&vec2[row * 3 + 2], rz * w2);
    }
}

// ---------------------------------------------------------------------------
// Node kernel: Gram-Schmidt + cross -> out[N,3,3]
// ---------------------------------------------------------------------------
__global__ void node_kernel(const float* __restrict__ vec1,
                            const float* __restrict__ vec2,
                            float* __restrict__ out, int n_nodes) {
    const int i = blockIdx.x * blockDim.x + threadIdx.x;
    if (i >= n_nodes) return;
    const float v1x = vec1[i * 3 + 0], v1y = vec1[i * 3 + 1], v1z = vec1[i * 3 + 2];
    const float v2x = vec2[i * 3 + 0], v2y = vec2[i * 3 + 1], v2z = vec2[i * 3 + 2];

    const float n1   = sqrtf(v1x * v1x + v1y * v1y + v1z * v1z);
    const float inv1 = 1.0f / fmaxf(n1, 1e-12f);
    const float e1x = v1x * inv1, e1y = v1y * inv1, e1z = v1z * inv1;

    const float d  = e1x * v2x + e1y * v2y + e1z * v2z;
    const float px = v2x - d * e1x, py = v2y - d * e1y, pz = v2z - d * e1z;
    const float np   = sqrtf(px * px + py * py + pz * pz);
    const float invp = 1.0f / fmaxf(np, 1e-12f);
    const float e2x = px * invp, e2y = py * invp, e2z = pz * invp;

    const float e3x = e1y * e2z - e1z * e2y;
    const float e3y = e1z * e2x - e1x * e2z;
    const float e3z = e1x * e2y - e1y * e2x;

    float* o = out + (size_t)i * 9;
    o[0] = e1x; o[1] = e2x; o[2] = e3x;
    o[3] = e1y; o[4] = e2y; o[5] = e3y;
    o[6] = e1z; o[7] = e2z; o[8] = e3z;
}

// ---------------------------------------------------------------------------
extern "C" void kernel_launch(void* const* d_in, const int* in_sizes, int n_in,
                              void* d_out, int out_size, void* d_ws, size_t ws_size,
                              hipStream_t stream) {
    const float* h          = (const float*)d_in[0];
    const float* x          = (const float*)d_in[1];
    const int*   edge_index = (const int*)  d_in[2];
    const float* edge_attr  = (const float*)d_in[3];
    const float* eW1  = (const float*)d_in[4];
    const float* eb1  = (const float*)d_in[5];
    const float* eW2  = (const float*)d_in[6];
    const float* eb2  = (const float*)d_in[7];
    const float* v1W1 = (const float*)d_in[8];
    const float* v1b1 = (const float*)d_in[9];
    const float* v1W2 = (const float*)d_in[10];
    const float* v1b2 = (const float*)d_in[11];
    const float* v2W1 = (const float*)d_in[12];
    const float* v2b1 = (const float*)d_in[13];
    const float* v2W2 = (const float*)d_in[14];
    const float* v2b2 = (const float*)d_in[15];

    const int n_edges = in_sizes[2] / 2;
    const int n_nodes = in_sizes[0] / HIDDEN;

    // ---- workspace layout ----
    char* ws = (char*)d_ws;
    _Float16* eW1s  = (_Float16*)ws; ws += (size_t)W1S_ELEMS * sizeof(_Float16);
    _Float16* eW2s  = (_Float16*)ws; ws += (size_t)W2S_ELEMS * sizeof(_Float16);
    _Float16* v1W1s = (_Float16*)ws; ws += (size_t)W2S_ELEMS * sizeof(_Float16);
    _Float16* v2W1s = (_Float16*)ws; ws += (size_t)W2S_ELEMS * sizeof(_Float16);
    float* vec1 = (float*)ws; ws += (size_t)n_nodes * 3 * sizeof(float);
    float* vec2 = (float*)ws; ws += (size_t)n_nodes * 3 * sizeof(float);

    const int nvec   = n_nodes * 3;
    const int prep_n = (W1S_ELEMS > nvec) ? W1S_ELEMS : nvec;
    prep_kernel<<<(prep_n + 255) / 256, 256, 0, stream>>>(
        eW1, eW2, v1W1, v2W1, eW1s, eW2s, v1W1s, v2W1s, vec1, vec2, nvec);

    const int tiles = (n_edges + 15) / 16;
    edge_kernel<<<(tiles + WAVES_PB - 1) / WAVES_PB, WAVES_PB * 32, 0, stream>>>(
        h, x, edge_index, edge_attr,
        eW1s, eb1, eW2s, eb2,
        v1W1s, v1b1, v1W2, v1b2,
        v2W1s, v2b1, v2W2, v2b2,
        vec1, vec2, n_edges);

    node_kernel<<<(n_nodes + 255) / 256, 256, 0, stream>>>(vec1, vec2, (float*)d_out, n_nodes);
}
